// ExpertAttention_49177375539835
// MI455X (gfx1250) — matrix operand order repaired and checked by reading, not compiled, not run on hardware
//
#include <hip/hip_runtime.h>

// ============================================================================
// ExpertAttention (Switch-LoRA MHA) for MI455X / gfx1250, wave32 WMMA + TDM.
//
// Math: out = MHA(x, common) + MHA(x, W_e + A_e@B_e), e = router(x); the
//       p_max/stop_grad(p_max) scale is exactly 1 in forward.
// Precision: f16 storage / WMMA f32 accumulate (v_wmma_f32_16x16x32_f16).
// Data movement: Tensor Data Mover (tensor_load_to_lds) double-buffered tile
//       pipeline; all weight matrices pre-transposed so every tile is a
//       linear 2D DMA.
// ============================================================================

typedef _Float16 h16;
typedef __attribute__((ext_vector_type(16))) _Float16 v16h;
typedef __attribute__((ext_vector_type(8)))  _Float16 v8h;
typedef __attribute__((ext_vector_type(8)))  float    v8f;
typedef __attribute__((ext_vector_type(4)))  unsigned int u32x4;
typedef __attribute__((ext_vector_type(8)))  int i32x8;
typedef __attribute__((ext_vector_type(4)))  int i32x4;

#define DEVI static __device__ __forceinline__

#if defined(__has_builtin)
#if __has_builtin(__builtin_amdgcn_tensor_load_to_lds) && \
    __has_builtin(__builtin_amdgcn_s_wait_tensorcnt)
#define USE_TDM 1
#endif
#endif
#ifndef USE_TDM
#define USE_TDM 0
#endif

constexpr int Bn = 64, Sn = 512, Dn = 768, Hn = 12, DHn = 64, Rn = 128;
constexpr float NEGMASK = -10000.0f;
constexpr float SM_SCALE = 0.125f;  // 1/sqrt(64)

DEVI v8h ldv8(const h16* p) { return *reinterpret_cast<const v8h*>(p); }
DEVI void stv8(h16* p, v8h v) { *reinterpret_cast<v8h*>(p) = v; }

DEVI v16h make16(v8h lo, v8h hi) {
  v16h a;
#pragma unroll
  for (int i = 0; i < 8; ++i) { a[i] = lo[i]; a[i + 8] = hi[i]; }
  return a;
}

DEVI v8f vzero8() {
  v8f z;
#pragma unroll
  for (int i = 0; i < 8; ++i) z[i] = 0.0f;
  return z;
}

// --- CDNA5 wave32 WMMA fragment loaders (CDNA5 ISA §7.12.2) -----
// A 16x32 f16, element (m,k)=base[m*ld+k]; lane L: m=L&15,
// lanes 0-15 hold K {0..7,16..23}, lanes 16-31 hold K {8..15,24..31}.
DEVI v16h load_a(const h16* base, int ld) {
  const int lane = threadIdx.x & 31;
  const int m = lane & 15, hi = lane >> 4;
  const h16* r = base + m * ld + hi * 8;
  return make16(ldv8(r), ldv8(r + 16));
}
// B 32x16 f16 from k-contiguous storage bt[n*ldt+k] (element (k,n)).
// lane L: n=L&15; lanes 0-15 hold K {0..7,16..23}, lanes 16-31 hold K {8..15,24..31}.
DEVI v16h load_b_t(const h16* bt, int ldt) {
  const int lane = threadIdx.x & 31;
  const int n = lane & 15, kb = (lane >> 4) * 8;
  const h16* r = bt + n * ldt + kb;
  return make16(ldv8(r), ldv8(r + 16));
}

DEVI v8f wmma_f16(v16h a, v16h b, v8f c) {
  return __builtin_amdgcn_wmma_f32_16x16x32_f16(false, a, false, b,
                                                (short)0, c, false, false);
}

#if USE_TDM
// --- Tensor Data Mover: 2D f16 tile DMA global->LDS (CDNA5 ISA*.md §8) --
// D# group0: count=1 | lds_addr | global_addr(57b) | type=2.
// D# group1: data_size=1(2B) | tensor_dim0/1 | tile_dim0/1 | dim0_stride.
// Dims/strides in data_size (element) units. Issued once per wave (EXEC
// ignored); completion tracked with TENSORcnt.
// This toolchain's builtin takes 6 args: (g0, g1, g2, g3, g4, cpol).
DEVI void tdm_load_2d(unsigned lds_off, const void* gaddr, unsigned td0,
                      unsigned td1, unsigned tile0, unsigned tile1,
                      unsigned stride0) {
  const unsigned long long ga = (unsigned long long)(size_t)gaddr;
  u32x4 g0;
  g0[0] = 1u;                                                // count=1
  g0[1] = lds_off;                                           // lds_addr
  g0[2] = (unsigned)ga;                                      // global[31:0]
  g0[3] = (unsigned)((ga >> 32) & 0x1FFFFFFu) | (2u << 30);  // global[56:32]|type=2
  i32x8 g1;
  g1[0] = (int)(1u << 16);                                   // data_size=2B
  g1[1] = (int)((td0 & 0xFFFFu) << 16);                      // tensor_dim0 lo
  g1[2] = (int)(((td0 >> 16) & 0xFFFFu) | ((td1 & 0xFFFFu) << 16));
  g1[3] = (int)(((td1 >> 16) & 0xFFFFu) | ((tile0 & 0xFFFFu) << 16));
  g1[4] = (int)(tile1 & 0xFFFFu);                            // tile_dim1 (dim2=0)
  g1[5] = (int)stride0;                                      // dim0_stride lo
  g1[6] = 0;                                                 // stride hi, dim1_stride=0
  g1[7] = 0;
  i32x4 z4;
  z4[0] = z4[1] = z4[2] = z4[3] = 0;
  i32x8 z8;
#pragma unroll
  for (int i = 0; i < 8; ++i) z8[i] = 0;
  __builtin_amdgcn_tensor_load_to_lds(g0, g1, z4, z4, z8, 0);
}
DEVI unsigned lds_off_of(const void* p) { return (unsigned)(size_t)p; }
#endif

// ============================================================================
// Generic WMMA GEMM:  out[M,N] = X[M,K] @ Wsel^T (+bias) (+addmat)
//   Wt storage is [N,K] (k-contiguous) so B tiles are linear 2D DMAs.
//   Wsel = routes ? (routes[row/512] ? W1 : W0) : W0  (per-batch expert).
//   outMode: 0 = outH[M,N]; 1 = outH[N,M] (weight fold); 2 = [B,H,DH,S]
//   (V layout for flash).  outF is f32 [M,N], optionally accumulating.
// Tile: 128(M) x 64(N) per 256-thread block (8 waves x 16 rows), BK=32,
// double-buffered TDM pipeline.
// ============================================================================
__global__ __launch_bounds__(256) void wmma_gemm(
    const h16* __restrict__ X, const h16* __restrict__ W0t,
    const h16* __restrict__ W1t, const float* __restrict__ bias0,
    const float* __restrict__ bias1, const float* __restrict__ addmat,
    const int* __restrict__ routes, int rows_per_batch,
    h16* __restrict__ outH, int outMode, float* __restrict__ outF, int accF,
    int M, int N, int K) {
  __shared__ __align__(16) h16 As[2][128 * 32];  // [row][k]
  __shared__ __align__(16) h16 Bs[2][64 * 32];   // [n][k]

  const int tid = threadIdx.x, wave = tid >> 5, lane = tid & 31;
  const int n0 = blockIdx.x * 64;
  const int mBase = blockIdx.y * 128;

  const h16* Wt = W0t;
  const float* bias = bias0;
  if (routes) {
    if (routes[mBase / rows_per_batch]) { Wt = W1t; bias = bias1; }
  }

  v8f acc[4];
#pragma unroll
  for (int j = 0; j < 4; ++j) acc[j] = vzero8();

  const int nIter = K / 32;

#if USE_TDM
  if (wave == 0) {
    tdm_load_2d(lds_off_of(&As[0][0]), X + (size_t)mBase * K, K, M, 32, 128, K);
    tdm_load_2d(lds_off_of(&Bs[0][0]), Wt + (size_t)n0 * K, K, N, 32, 64, K);
  }
  for (int it = 0; it < nIter; ++it) {
    const int buf = it & 1;
    if (wave == 0) {
      if (it + 1 < nIter) {
        const int k0 = (it + 1) * 32;
        tdm_load_2d(lds_off_of(&As[buf ^ 1][0]), X + (size_t)mBase * K + k0, K,
                    M, 32, 128, K);
        tdm_load_2d(lds_off_of(&Bs[buf ^ 1][0]), Wt + (size_t)n0 * K + k0, K,
                    N, 32, 64, K);
        __builtin_amdgcn_s_wait_tensorcnt(2);  // current pair landed
      } else {
        __builtin_amdgcn_s_wait_tensorcnt(0);
      }
    }
    __syncthreads();
    v16h af = load_a(&As[buf][(wave * 16) * 32], 32);
    v16h bf[4];
#pragma unroll
    for (int j = 0; j < 4; ++j) bf[j] = load_b_t(&Bs[buf][(j * 16) * 32], 32);
#pragma unroll
    for (int j = 0; j < 4; ++j) acc[j] = wmma_f16(af, bf[j], acc[j]);
    __syncthreads();  // safe to overwrite this buffer with load it+2
  }
#else
  const int rA = tid >> 1, cA = (tid & 1) * 16;  // 128 x 32 (16 h/thr)
  const int rB = tid >> 2, cB = (tid & 3) * 8;   // 64 x 32 (8 h/thr)
  for (int it = 0; it < nIter; ++it) {
    const int k0 = it * 32;
    v8h a0 = ldv8(X + (size_t)(mBase + rA) * K + k0 + cA);
    v8h a1 = ldv8(X + (size_t)(mBase + rA) * K + k0 + cA + 8);
    v8h wv = ldv8(Wt + (size_t)(n0 + rB) * K + k0 + cB);
    __syncthreads();
    stv8(&As[0][rA * 32 + cA], a0);
    stv8(&As[0][rA * 32 + cA + 8], a1);
    stv8(&Bs[0][rB * 32 + cB], wv);
    __syncthreads();
    v16h af = load_a(&As[0][(wave * 16) * 32], 32);
    v16h bf[4];
#pragma unroll
    for (int j = 0; j < 4; ++j) bf[j] = load_b_t(&Bs[0][(j * 16) * 32], 32);
#pragma unroll
    for (int j = 0; j < 4; ++j) acc[j] = wmma_f16(af, bf[j], acc[j]);
  }
#endif

  // Epilogue: C layout lane L, vgpr r -> (m = r + (L>=16)*8, n = L&15).
  const int mTop = mBase + wave * 16 + (lane >> 4) * 8;
  const int nLo = n0 + (lane & 15);
#pragma unroll
  for (int j = 0; j < 4; ++j) {
    const int n = nLo + j * 16;
    const float bv = bias ? bias[n] : 0.0f;
#pragma unroll
    for (int r = 0; r < 8; ++r) {
      const int mm = mTop + r;
      float v = acc[j][r] + bv;
      if (addmat) v += addmat[(size_t)mm * N + n];
      if (outH) {
        size_t oidx;
        if (outMode == 0) {
          oidx = (size_t)mm * N + n;
        } else if (outMode == 1) {        // transposed weight store [N,M]
          oidx = (size_t)n * M + mm;
        } else {                          // V head-transposed [B,H,DH,S]
          const int bb = mm / Sn, ss = mm % Sn, hh = n / DHn, dd = n % DHn;
          oidx = (((size_t)bb * Hn + hh) * DHn + dd) * Sn + ss;
        }
        outH[oidx] = (h16)v;
      }
      if (outF) {
        const size_t idx = (size_t)mm * N + n;
        if (accF) outF[idx] += v; else outF[idx] = v;
      }
    }
  }
}

// ============================================================================
// Flash attention, per (b, h, 128-row q-tile); 8 waves x 16 q-rows.
// K tiles ([krow][dh]) and V tiles ([dh][krow], from the [B,H,DH,S] buffer)
// are TDM double-buffered; online softmax in f32; P re-laid out via per-wave
// LDS bounce into an A-fragment.
// ============================================================================
__global__ __launch_bounds__(256) void flash_attn(
    const h16* __restrict__ Q, const h16* __restrict__ Kh,
    const h16* __restrict__ Vt_g, const float* __restrict__ mask,
    h16* __restrict__ ctx) {
  __shared__ __align__(16) h16 Ks[2][32 * 64];   // [krow][dh]
  __shared__ __align__(16) h16 Vt[2][64 * 32];   // [dh][krow]
  __shared__ __align__(16) h16 Ps[8][16 * 32];   // per-wave P tile [m][k]

  const int tid = threadIdx.x, wave = tid >> 5, lane = tid & 31;
  const int h = blockIdx.y, b = blockIdx.z;
  const int q0 = blockIdx.x * 128 + wave * 16;
  const size_t headBase = (size_t)b * Sn * Dn + (size_t)h * DHn;
  const h16* Vhead = Vt_g + ((size_t)b * Hn + h) * DHn * Sn;  // [DH][S]

  v16h qa[2];
  {
    const int m = lane & 15, hi = lane >> 4;
    const h16* qp = Q + headBase + (size_t)(q0 + m) * Dn;
#pragma unroll
    for (int c = 0; c < 2; ++c) {
      const h16* r = qp + c * 32 + hi * 8;
      qa[c] = make16(ldv8(r), ldv8(r + 16));
    }
  }

  v8f O[4];
#pragma unroll
  for (int j = 0; j < 4; ++j) O[j] = vzero8();
  float mrow[8], lrow[8];
#pragma unroll
  for (int r = 0; r < 8; ++r) { mrow[r] = -3.0e38f; lrow[r] = 0.0f; }

  const int nIter = Sn / 32;
#if USE_TDM
  if (wave == 0) {
    tdm_load_2d(lds_off_of(&Ks[0][0]), Kh + headBase, Dn, Bn * Sn, 64, 32, Dn);
    tdm_load_2d(lds_off_of(&Vt[0][0]), Vhead, Sn, Bn * Hn * DHn, 32, 64, Sn);
  }
#else
  const int rL = tid >> 3, cL = (tid & 7) * 8;
#endif

  for (int it = 0; it < nIter; ++it) {
    const int buf = it & 1;
    const int kt = it * 32;
#if USE_TDM
    if (wave == 0) {
      if (it + 1 < nIter) {
        const int kn = kt + 32;
        tdm_load_2d(lds_off_of(&Ks[buf ^ 1][0]),
                    Kh + headBase + (size_t)kn * Dn, Dn, Bn * Sn, 64, 32, Dn);
        tdm_load_2d(lds_off_of(&Vt[buf ^ 1][0]), Vhead + kn, Sn,
                    Bn * Hn * DHn, 32, 64, Sn);
        __builtin_amdgcn_s_wait_tensorcnt(2);
      } else {
        __builtin_amdgcn_s_wait_tensorcnt(0);
      }
    }
    __syncthreads();
#else
    v8h kv = ldv8(Kh + headBase + (size_t)(kt + rL) * Dn + cL);
    __syncthreads();
    stv8(&Ks[0][rL * 64 + cL], kv);
    {  // V tile [dh][krow] straight from [DH][S] layout: 64 rows x 32 cols
      const int rV = tid >> 2, cV = (tid & 3) * 8;
      v8h vr = ldv8(Vhead + (size_t)rV * Sn + kt + cV);
      stv8(&Vt[0][rV * 32 + cV], vr);
    }
    __syncthreads();
#endif

    // --- scores: two 16x16 fragments over the 32 k-columns ------------------
    v8f sc[2];
#pragma unroll
    for (int nf = 0; nf < 2; ++nf) {
      v8f c = vzero8();
      v16h kb0, kb1;
      {
        const int n = lane & 15, kb = (lane >> 4) * 8;
        const h16* r0 = &Ks[buf][(nf * 16 + n) * 64 + 0 + kb];
        const h16* r1 = &Ks[buf][(nf * 16 + n) * 64 + 32 + kb];
        kb0 = make16(ldv8(r0), ldv8(r0 + 16));
        kb1 = make16(ldv8(r1), ldv8(r1 + 16));
      }
      c = wmma_f16(qa[0], kb0, c);
      c = wmma_f16(qa[1], kb1, c);
      const int col = kt + nf * 16 + (lane & 15);
      const float madd = (1.0f - mask[(size_t)b * Sn + col]) * NEGMASK;
#pragma unroll
      for (int r = 0; r < 8; ++r) c[r] = c[r] * SM_SCALE + madd;
      sc[nf] = c;
    }

    // --- online softmax (rows live across the 16 lanes of each half) -------
    float alpha[8];
#pragma unroll
    for (int r = 0; r < 8; ++r) {
      float v = fmaxf(sc[0][r], sc[1][r]);
#pragma unroll
      for (int off = 1; off < 16; off <<= 1)
        v = fmaxf(v, __shfl_xor(v, off, 32));
      const float mn = fmaxf(mrow[r], v);
      alpha[r] = __expf(mrow[r] - mn);
      mrow[r] = mn;
    }
#pragma unroll
    for (int r = 0; r < 8; ++r) {
      const float p0 = __expf(sc[0][r] - mrow[r]);
      const float p1 = __expf(sc[1][r] - mrow[r]);
      sc[0][r] = p0; sc[1][r] = p1;
      float v = p0 + p1;
#pragma unroll
      for (int off = 1; off < 16; off <<= 1) v += __shfl_xor(v, off, 32);
      lrow[r] = lrow[r] * alpha[r] + v;
    }

    // --- P: C-layout -> per-wave LDS -> A-layout fragment -------------------
    h16* ps = Ps[wave];
    {
      const int mh = (lane >> 4) * 8, n = lane & 15;
#pragma unroll
      for (int nf = 0; nf < 2; ++nf)
#pragma unroll
        for (int r = 0; r < 8; ++r)
          ps[(mh + r) * 32 + nf * 16 + n] = (h16)sc[nf][r];
    }
#pragma unroll
    for (int j = 0; j < 4; ++j)
#pragma unroll
      for (int r = 0; r < 8; ++r) O[j][r] *= alpha[r];

    v16h pa = load_a(ps, 32);
    v16h vb[4];
#pragma unroll
    for (int j = 0; j < 4; ++j) vb[j] = load_b_t(&Vt[buf][(j * 16) * 32], 32);
#pragma unroll
    for (int j = 0; j < 4; ++j) O[j] = wmma_f16(pa, vb[j], O[j]);
    __syncthreads();  // buffer may be overwritten by the it+2 DMA
  }

  // --- normalize and store ctx (f16, [B*S, 768] + head column offset) ------
  {
    const int mh = (lane >> 4) * 8, n = lane & 15;
#pragma unroll
    for (int r = 0; r < 8; ++r) {
      const float inv = 1.0f / lrow[r];
      h16* cp = ctx + headBase + (size_t)(q0 + mh + r) * Dn;
#pragma unroll
      for (int j = 0; j < 4; ++j) cp[j * 16 + n] = (h16)(O[j][r] * inv);
    }
  }
}

// ============================================================================
// Small helper kernels
// ============================================================================
__global__ void f32_to_f16(const float* __restrict__ s, h16* __restrict__ d,
                           int n) {
  int i = blockIdx.x * blockDim.x + threadIdx.x;
  const int stride = gridDim.x * blockDim.x;
  for (; i < n; i += stride) d[i] = (h16)s[i];
}

// f32 [K,N] row-major -> f16 [N,K] (k-contiguous for TDM B tiles)
__global__ void f32_to_f16_T(const float* __restrict__ s, h16* __restrict__ d,
                             int Kd, int Nd) {
  const int i = blockIdx.x * blockDim.x + threadIdx.x;
  if (i >= Kd * Nd) return;
  const int k = i / Nd, n = i % Nd;
  d[(size_t)n * Kd + k] = (h16)s[i];
}

__global__ void pool_kernel(const float* __restrict__ x,
                            float* __restrict__ pooled) {
  const int i = blockIdx.x * blockDim.x + threadIdx.x;  // B*D
  if (i >= Bn * Dn) return;
  const int b = i / Dn, d = i % Dn;
  const float* p = x + (size_t)b * Sn * Dn + d;
  float s = 0.0f;
  for (int j = 0; j < Sn; ++j) s += p[(size_t)j * Dn];
  pooled[i] = s * (1.0f / (float)Sn);
}

__global__ void router_kernel(const float* __restrict__ pooled,
                              const float* __restrict__ encW,
                              const float* __restrict__ encb,
                              const float* __restrict__ swW,
                              const float* __restrict__ swb,
                              int* __restrict__ routes) {
  const int b = blockIdx.x, t = threadIdx.x;  // 128 threads
  __shared__ float henc[Rn];
  float acc = encb[t];
  for (int d = 0; d < Dn; ++d) acc += pooled[b * Dn + d] * encW[d * Rn + t];
  henc[t] = acc;
  __syncthreads();
  if (t == 0) {
    float l0 = swb[0], l1 = swb[1];
    for (int r = 0; r < Rn; ++r) {
      l0 += henc[r] * swW[r * 2 + 0];
      l1 += henc[r] * swW[r * 2 + 1];
    }
    routes[b] = (l1 > l0) ? 1 : 0;  // argmax (softmax monotone; ties -> 0)
  }
}

// ============================================================================
// Host orchestration
// ============================================================================
extern "C" void kernel_launch(void* const* d_in, const int* in_sizes, int n_in,
                              void* d_out, int out_size, void* d_ws,
                              size_t ws_size, hipStream_t stream) {
  (void)in_sizes; (void)n_in; (void)out_size; (void)ws_size;

  // setup_inputs() flattening order:
  // 0:hidden_states 1:attention_mask 2:enc_W 3:enc_b 4:sw_W 5:sw_b
  // 6..13: common {Wq,bq,Wk,bk,Wv,bv,Wo,bo}
  // 14..29 / 30..45: expert e {Wq,bq,Aq,Bq, Wk,bk,Ak,Bk, Wv,bv,Av,Bv,
  //                            Wo,bo,Ao,Bo}
  const float* x    = (const float*)d_in[0];
  const float* msk  = (const float*)d_in[1];
  const float* encW = (const float*)d_in[2];
  const float* encb = (const float*)d_in[3];
  const float* swW  = (const float*)d_in[4];
  const float* swb  = (const float*)d_in[5];
  const float *cW[4], *cb[4];
  for (int p = 0; p < 4; ++p) {
    cW[p] = (const float*)d_in[6 + 2 * p];
    cb[p] = (const float*)d_in[7 + 2 * p];
  }
  const float *eW[2][4], *eb[2][4], *eA[2][4], *eB[2][4];
  for (int e = 0; e < 2; ++e)
    for (int p = 0; p < 4; ++p) {
      const int base = 14 + e * 16 + p * 4;
      eW[e][p] = (const float*)d_in[base + 0];
      eb[e][p] = (const float*)d_in[base + 1];
      eA[e][p] = (const float*)d_in[base + 2];
      eB[e][p] = (const float*)d_in[base + 3];
    }
  float* out = (float*)d_out;

  // ---- workspace carve-up -------------------------------------------------
  char* wp = (char*)d_ws;
  auto take = [&](size_t bytes) -> void* {
    void* p = (void*)wp;
    wp += (bytes + 255) & ~(size_t)255;
    return p;
  };
  const size_t nAct = (size_t)Bn * Sn * Dn;
  h16* xh   = (h16*)take(nAct * 2);
  h16* q16  = (h16*)take(nAct * 2);
  h16* k16  = (h16*)take(nAct * 2);
  h16* v16t = (h16*)take(nAct * 2);   // [B,H,DH,S]
  h16* c16  = (h16*)take(nAct * 2);
  h16* cWt[4];
  for (int p = 0; p < 4; ++p) cWt[p] = (h16*)take((size_t)Dn * Dn * 2);
  h16 *eAh[2][4], *eBt[2][4], *WeffT[2][4];
  for (int e = 0; e < 2; ++e)
    for (int p = 0; p < 4; ++p) {
      eAh[e][p]   = (h16*)take((size_t)Dn * Rn * 2);  // [D,R] row-major
      eBt[e][p]   = (h16*)take((size_t)Rn * Dn * 2);  // [N=768, K=128]
      WeffT[e][p] = (h16*)take((size_t)Dn * Dn * 2);  // [N,K] transposed
    }
  float* pooled = (float*)take((size_t)Bn * Dn * 4);
  int* routes   = (int*)take((size_t)Bn * 4);

  auto cvt = [&](const float* s, h16* d, size_t n) {
    int blocks = (int)((n + 255) / 256);
    if (blocks > 4096) blocks = 4096;
    f32_to_f16<<<blocks, 256, 0, stream>>>(s, d, (int)n);
  };
  auto cvtT = [&](const float* s, h16* d, int Kd, int Nd) {
    f32_to_f16_T<<<(Kd * Nd + 255) / 256, 256, 0, stream>>>(s, d, Kd, Nd);
  };
  auto gemm = [&](const h16* X, const h16* W0t, const h16* W1t,
                  const float* b0, const float* b1, const float* addm,
                  const int* rts, h16* oh, int omode, float* of, int acc,
                  int M, int N, int K) {
    dim3 grid(N / 64, M / 128);
    wmma_gemm<<<grid, 256, 0, stream>>>(X, W0t, W1t, b0, b1, addm, rts, Sn,
                                        oh, omode, of, acc, M, N, K);
  };

  // ---- stage 0: conversions / weight transposes ---------------------------
  cvt(x, xh, nAct);
  for (int p = 0; p < 4; ++p) cvtT(cW[p], cWt[p], Dn, Dn);
  for (int e = 0; e < 2; ++e)
    for (int p = 0; p < 4; ++p) {
      cvt(eA[e][p], eAh[e][p], (size_t)Dn * Rn);
      cvtT(eB[e][p], eBt[e][p], Rn, Dn);
    }

  // ---- stage 1: router ----------------------------------------------------
  pool_kernel<<<(Bn * Dn + 255) / 256, 256, 0, stream>>>(x, pooled);
  router_kernel<<<Bn, Rn, 0, stream>>>(pooled, encW, encb, swW, swb, routes);

  // ---- stage 2: fold LoRA -> W_eff^T = (W + A@B)^T (f16, [N,K]) -----------
  for (int e = 0; e < 2; ++e)
    for (int p = 0; p < 4; ++p)
      gemm(eAh[e][p], eBt[e][p], nullptr, nullptr, nullptr, eW[e][p], nullptr,
           WeffT[e][p], 1, nullptr, 0, Dn, Dn, Rn);

  const int M = Bn * Sn;
  dim3 agrid(Sn / 128, Hn, Bn);

  // ---- pass A: common MHA -> d_out (overwrite) ----------------------------
  gemm(xh, cWt[0], nullptr, cb[0], nullptr, nullptr, nullptr, q16, 0, nullptr, 0, M, Dn, Dn);
  gemm(xh, cWt[1], nullptr, cb[1], nullptr, nullptr, nullptr, k16, 0, nullptr, 0, M, Dn, Dn);
  gemm(xh, cWt[2], nullptr, cb[2], nullptr, nullptr, nullptr, v16t, 2, nullptr, 0, M, Dn, Dn);
  flash_attn<<<agrid, 256, 0, stream>>>(q16, k16, v16t, msk, c16);
  gemm(c16, cWt[3], nullptr, cb[3], nullptr, nullptr, nullptr, nullptr, 0, out, 0, M, Dn, Dn);

  // ---- pass B: routed expert MHA (folded weights) -> d_out (+=) -----------
  gemm(xh, WeffT[0][0], WeffT[1][0], eb[0][0], eb[1][0], nullptr, routes, q16, 0, nullptr, 0, M, Dn, Dn);
  gemm(xh, WeffT[0][1], WeffT[1][1], eb[0][1], eb[1][1], nullptr, routes, k16, 0, nullptr, 0, M, Dn, Dn);
  gemm(xh, WeffT[0][2], WeffT[1][2], eb[0][2], eb[1][2], nullptr, routes, v16t, 2, nullptr, 0, M, Dn, Dn);
  flash_attn<<<agrid, 256, 0, stream>>>(q16, k16, v16t, msk, c16);
  gemm(c16, WeffT[0][3], WeffT[1][3], eb[0][3], eb[1][3], nullptr, routes, nullptr, 0, out, 1, M, Dn, Dn);
}
